// q_Mlp_30562987278756
// MI455X (gfx1250) — 8-bit matrix operand order repaired and checked by reading, compiled, not run on hardware
//
#include <hip/hip_runtime.h>
#include <hip/hip_bf16.h>
#include <math.h>

typedef __attribute__((ext_vector_type(8))) int v8i;

#define QMAXF 127.0f

__device__ __forceinline__ float clampf(float v, float lo, float hi) {
    return fminf(fmaxf(v, lo), hi);
}

__device__ __forceinline__ unsigned lds_off32(const void* p) {
    return (unsigned)(uintptr_t)(__attribute__((address_space(3))) const void*)p;
}

__global__ void init_scalars_kernel(unsigned* amax) {
    if (threadIdx.x < 2) amax[threadIdx.x] = 0u;
}

__global__ __launch_bounds__(256)
void quant_w_kernel(const float* __restrict__ W, const float* __restrict__ b,
                    const float* __restrict__ prevPtr, const unsigned* __restrict__ amaxU,
                    signed char* __restrict__ Wq, int* __restrict__ bInt,
                    float* __restrict__ bScale, float* __restrict__ scaleOut,
                    float* __restrict__ actScaleOut, int IN)
{
    __shared__ float red[256];
    __shared__ float s_ws;
    const int o = blockIdx.x;
    const float* row = W + (size_t)o * IN;

    float m = 0.f;
    for (int i = threadIdx.x; i < IN; i += 256) m = fmaxf(m, fabsf(row[i]));
    red[threadIdx.x] = m;
    __syncthreads();
    for (int s = 128; s > 0; s >>= 1) {
        if (threadIdx.x < s) red[threadIdx.x] = fmaxf(red[threadIdx.x], red[threadIdx.x + s]);
        __syncthreads();
    }
    if (threadIdx.x == 0) {
        float prev = prevPtr ? prevPtr[0]
                             : (__uint_as_float(amaxU[0]) / QMAXF + 1e-8f);
        float ws = red[0] / QMAXF + 1e-8f;
        s_ws = ws;
        float bs = ws * prev;
        bScale[o] = bs;
        bInt[o]   = (int)rintf(b[o] / bs);
        if (scaleOut) scaleOut[o] = ws;
        if (actScaleOut && o == 0) actScaleOut[0] = prev;
    }
    __syncthreads();
    const float ws = s_ws;
    signed char* qrow = Wq + (size_t)o * IN;
    for (int i = threadIdx.x; i < IN; i += 256) {
        float q = rintf(row[i] / ws);
        qrow[i] = (signed char)clampf(q, -128.f, 127.f);
    }
}

__global__ void quant_x_kernel(const float* __restrict__ x,
                               const float* __restrict__ prevPtr,
                               signed char* __restrict__ x8, size_t n)
{
    size_t i = (size_t)blockIdx.x * 256 + threadIdx.x;
    if (i >= n) return;
    float prev = prevPtr[0];
    x8[i] = (signed char)clampf(rintf(x[i] / prev), -128.f, 127.f);
}

template <typename TOut>
__global__ __launch_bounds__(256)
void gemm_i8_wmma_kernel(const signed char* __restrict__ A8,
                         const signed char* __restrict__ B8,
                         const int*  __restrict__ bInt,
                         const float* __restrict__ bScale,
                         TOut* __restrict__ Out,
                         int M, int N, int K,
                         unsigned* amaxOut)
{
    constexpr int BSTRIDE = 80;
    __shared__ signed char Btile[2][64 * BSTRIDE];
    __shared__ float red[256];

    const int tid    = threadIdx.x;
    const int lane   = tid & 31;
    const int l16    = lane & 15;
    const int half   = lane >> 4;
    const int waveId = tid >> 5;

    const int  mTile  = blockIdx.x * 8 + waveId;
    const bool active = __builtin_amdgcn_readfirstlane(mTile) < (M >> 4);
    const int  mBase  = mTile << 4;
    const int  nBase  = blockIdx.y << 6;

    const int sCol  = tid >> 2;
    const int sKoff = (tid & 3) << 4;
    const signed char* sGlob = B8 + (size_t)(nBase + sCol) * K + sKoff;
    const unsigned sLds0 = lds_off32(&Btile[0][0]) + sCol * BSTRIDE + sKoff;
    const unsigned sLds1 = lds_off32(&Btile[1][0]) + sCol * BSTRIDE + sKoff;

    v8i acc[4];
#pragma unroll
    for (int t = 0; t < 4; ++t) {
        v8i z = {0, 0, 0, 0, 0, 0, 0, 0};
        acc[t] = z;
    }

    const signed char* aRow = A8 + (size_t)(mBase + l16) * K;
    float localMax = 0.f;

    asm volatile("global_load_async_to_lds_b128 %0, %1, off"
                 :: "v"(sLds0), "v"(sGlob) : "memory");

    if (active) {
        int cur = 0;
        for (int k0 = 0; k0 < K; k0 += 64, cur ^= 1) {
            asm volatile("s_wait_asynccnt 0" ::: "memory");
            __syncthreads();
            if (k0 + 64 < K) {
                unsigned dst = cur ? sLds0 : sLds1;
                asm volatile("global_load_async_to_lds_b128 %0, %1, off"
                             :: "v"(dst), "v"(sGlob + k0 + 64) : "memory");
            }

            v8i a;
#pragma unroll
            for (int j = 0; j < 4; ++j) {
                int2 d = *(const int2*)(aRow + k0 + (j >> 1) * 32 + half * 16 + (j & 1) * 8);
                a[2 * j]     = d.x;
                a[2 * j + 1] = d.y;
            }
            if (k0 + 64 < K) __builtin_prefetch(aRow + k0 + 64, 0, 1);

            const signed char* bBuf = &Btile[cur][0];
#pragma unroll
            for (int t = 0; t < 4; ++t) {
                const signed char* bRow = bBuf + (t * 16 + l16) * BSTRIDE + half * 16;
                int4 p0 = *(const int4*)(bRow);
                int4 p1 = *(const int4*)(bRow + 32);
                v8i bm;
                bm[0] = p0.x; bm[1] = p0.y; bm[2] = p0.z; bm[3] = p0.w;
                bm[4] = p1.x; bm[5] = p1.y; bm[6] = p1.z; bm[7] = p1.w;
                acc[t] = __builtin_amdgcn_wmma_i32_16x16x64_iu8(
                     true, a,  true, bm, acc[t],
                     false,  false);
            }
        }

#pragma unroll
        for (int t = 0; t < 4; ++t) {
            int n = nBase + t * 16 + l16;
            int   bi = bInt[n];
            float bs = bScale[n];
#pragma unroll
            for (int r = 0; r < 8; ++r) {
                int m = mBase + (half << 3) + r;
                float v = (float)(acc[t][r] + bi) * bs;
                Out[(size_t)m * N + n] = (TOut)v;
                localMax = fmaxf(localMax, fabsf(v));
            }
        }
    } else {
        int cur = 0;
        for (int k0 = 0; k0 < K; k0 += 64, cur ^= 1) {
            asm volatile("s_wait_asynccnt 0" ::: "memory");
            __syncthreads();
            if (k0 + 64 < K) {
                unsigned dst = cur ? sLds0 : sLds1;
                asm volatile("global_load_async_to_lds_b128 %0, %1, off"
                             :: "v"(dst), "v"(sGlob + k0 + 64) : "memory");
            }
        }
    }

    if (amaxOut) {
        red[tid] = localMax;
        __syncthreads();
        for (int s = 128; s > 0; s >>= 1) {
            if (tid < s) red[tid] = fmaxf(red[tid], red[tid + s]);
            __syncthreads();
        }
        if (tid == 0) atomicMax(amaxOut, __float_as_uint(red[0]));
    }
}

__global__ __launch_bounds__(256)
void act_gelu_kernel(_Float16* __restrict__ Hb, const unsigned* __restrict__ amax1,
                     unsigned* __restrict__ amax2, size_t n)
{
    __shared__ float red[256];
    const float s1 = __uint_as_float(amax1[0]) / QMAXF + 1e-8f;
    size_t i = (size_t)blockIdx.x * 256 + threadIdx.x;
    float g = 0.f;
    if (i < n) {
        float h  = (float)Hb[i];
        float hq = clampf(rintf(h / s1), -128.f, 127.f) * s1;
        g = 0.5f * hq * (1.f + erff(hq * 0.70710678118654752f));
        Hb[i] = (_Float16)g;
    }
    red[threadIdx.x] = fabsf(g);
    __syncthreads();
    for (int s = 128; s > 0; s >>= 1) {
        if (threadIdx.x < s) red[threadIdx.x] = fmaxf(red[threadIdx.x], red[threadIdx.x + s]);
        __syncthreads();
    }
    if (threadIdx.x == 0) atomicMax(amax2, __float_as_uint(red[0]));
}

__global__ void quant_g_kernel(const _Float16* __restrict__ Hb,
                               const unsigned* __restrict__ amax2,
                               signed char* __restrict__ G8, size_t n)
{
    size_t i = (size_t)blockIdx.x * 256 + threadIdx.x;
    if (i >= n) return;
    float s2 = __uint_as_float(amax2[0]) / QMAXF + 1e-8f;
    G8[i] = (signed char)clampf(rintf((float)Hb[i] / s2), -128.f, 127.f);
}

extern "C" void kernel_launch(void* const* d_in, const int* in_sizes, int n_in,
                              void* d_out, int out_size, void* d_ws, size_t ws_size,
                              hipStream_t stream) {
    (void)in_sizes; (void)n_in; (void)out_size; (void)ws_size;
    const float* x    = (const float*)d_in[0];
    const float* W1   = (const float*)d_in[1];
    const float* b1   = (const float*)d_in[2];
    const float* W2   = (const float*)d_in[3];
    const float* b2   = (const float*)d_in[4];
    const float* prev = (const float*)d_in[5];

    const int C = 768, H = 3072;
    const int M = 64 * 197;

    float* out = (float*)d_out;

    char* ws = (char*)d_ws;
    size_t off = 0;
    auto take = [&](size_t bytes) -> char* {
        char* p = ws + off;
        off = (off + bytes + 255) & ~(size_t)255;
        return p;
    };
    unsigned*    amax = (unsigned*)   take(256);
    signed char* X8   = (signed char*)take((size_t)M * C);
    signed char* W1q  = (signed char*)take((size_t)H * C);
    signed char* W2q  = (signed char*)take((size_t)C * H);
    int*         b1i  = (int*)        take((size_t)H * 4);
    int*         b2i  = (int*)        take((size_t)C * 4);
    float*       bs1  = (float*)      take((size_t)H * 4);
    float*       bs2  = (float*)      take((size_t)C * 4);
    signed char* G8   = (signed char*)take((size_t)M * H);
    _Float16*    Hbuf = (_Float16*)   take((size_t)M * H * 2);

    init_scalars_kernel<<<1, 64, 0, stream>>>(amax);

    quant_w_kernel<<<H, 256, 0, stream>>>(W1, b1, prev, nullptr,
                                          W1q, b1i, bs1, nullptr, nullptr, C);

    {
        size_t n = (size_t)M * C;
        quant_x_kernel<<<(unsigned)((n + 255) / 256), 256, 0, stream>>>(x, prev, X8, n);
    }

    const unsigned gridM = (unsigned)(((M / 16) + 7) / 8);

    gemm_i8_wmma_kernel<_Float16><<<dim3(gridM, H / 64), 256, 0, stream>>>(
        X8, W1q, b1i, bs1, Hbuf, M, H, C, amax + 0);

    {
        size_t n = (size_t)M * H;
        act_gelu_kernel<<<(unsigned)((n + 255) / 256), 256, 0, stream>>>(
            Hbuf, amax + 0, amax + 1, n);
    }

    quant_w_kernel<<<C, 256, 0, stream>>>(W2, b2, nullptr, amax + 1,
                                          W2q, b2i, bs2,
                                          out + (size_t)M * C + 1,
                                          out + (size_t)M * C,
                                          H);

    {
        size_t n = (size_t)M * H;
        quant_g_kernel<<<(unsigned)((n + 255) / 256), 256, 0, stream>>>(
            Hbuf, amax + 1, G8, n);
    }

    gemm_i8_wmma_kernel<float><<<dim3(gridM, C / 64), 256, 0, stream>>>(
        G8, W2q, b2i, bs2, out, M, C, H, nullptr);
}
